// GNNWithInjectedTemp_53712861004551
// MI455X (gfx1250) — matrix operand order repaired and checked by reading, not compiled, not run on hardware
//
#include <hip/hip_runtime.h>
#include <hip/hip_bf16.h>

// ---------------------------------------------------------------------------
// GAT (GNNWithInjectedTemp) forward for MI455X / gfx1250 (wave32, WMMA).
//
// Input pointer layout assumption (setup_inputs() dict order, params flattened
// in insertion order as written in make_params):
//   0:x(N,7) 1:ini(N,1) 2:x_branch(B,32) 3:sample_idx(B) 4:edge_index(2,E)
//   5:edge_attr(E,3)
//   6:tW1(26,96) 7:tb1 8:tW2(96,64) 9:tb2 10:tlw 11:tlb
//   12+5l: Wn_l   13+5l: bn_l   14+5l: We_l(3,64) 15+5l: be_l 16+5l: attn_l
//   37+3l: gnw_l  38+3l: gnb_l  39+3l: gnms_l     (l = 0..4)
//   52:rW1(64,32) 53:rb1 54:rW2(32,1) 55:rb2
// ---------------------------------------------------------------------------

#define NN 50000
#define NE 800000
#define NB 4096

typedef _Float16 f16;
typedef __attribute__((ext_vector_type(2)))  _Float16 f16x2;
typedef __attribute__((ext_vector_type(8)))  _Float16 f16x8;
typedef __attribute__((ext_vector_type(16))) _Float16 v16h;
typedef __attribute__((ext_vector_type(8)))  float    v8f;

__device__ __forceinline__ float leakyf(float x, float s) { return x >= 0.f ? x : s * x; }

// float atomic max via int punning (monotone encodings); v>0 -> signed max,
// else unsigned min. Valid with -inf initialization.
__device__ __forceinline__ void atomicMaxF(float* a, float v) {
  if (v > 0.f) atomicMax((int*)a, __float_as_int(v));
  else         atomicMin((unsigned int*)a, (unsigned int)__float_as_int(v));
}

// ---------------- branch-temperature MLP ----------------
__global__ void k_bmlp1(const float* __restrict__ xb, const float* __restrict__ W,
                        const float* __restrict__ b, float* __restrict__ t1) {
  int i = blockIdx.x * blockDim.x + threadIdx.x;
  if (i >= NB * 96) return;
  int row = i / 96, c = i % 96;
  float acc = b[c];
  const float* xr = xb + row * 32 + 6;           // x_branch[:, 6:], 26 features
#pragma unroll
  for (int k = 0; k < 26; ++k) acc += xr[k] * W[k * 96 + c];
  t1[i] = leakyf(acc, 0.01f);
}

__global__ void k_bmlp2(const float* __restrict__ t1, const float* __restrict__ W,
                        const float* __restrict__ b, float* __restrict__ t2) {
  int i = blockIdx.x * blockDim.x + threadIdx.x;
  if (i >= NB * 64) return;
  int row = i / 64, c = i % 64;
  float acc = b[c];
  const float* tr = t1 + row * 96;
#pragma unroll 8
  for (int k = 0; k < 96; ++k) acc += tr[k] * W[k * 64 + c];
  t2[i] = acc;
}

__global__ void k_bln(float* __restrict__ t, const float* __restrict__ w,
                      const float* __restrict__ b) {
  int row = blockIdx.x * blockDim.x + threadIdx.x;
  if (row >= NB) return;
  float* tr = t + row * 64;
  float m = 0.f;
#pragma unroll 8
  for (int k = 0; k < 64; ++k) m += tr[k];
  m *= (1.f / 64.f);
  float v = 0.f;
#pragma unroll 8
  for (int k = 0; k < 64; ++k) { float d = tr[k] - m; v += d * d; }
  v *= (1.f / 64.f);
  float inv = rsqrtf(v + 1e-5f);
#pragma unroll 8
  for (int k = 0; k < 64; ++k)
    tr[k] = leakyf((tr[k] - m) * inv * w[k] + b[k], 0.01f);
}

// ---------------- input feature assembly: hAh is (N, 96) f16, zero-padded ----------------
__global__ void k_fill_h0(const float* __restrict__ x, const float* __restrict__ ini,
                          f16* __restrict__ hAh) {
  int i = blockIdx.x * blockDim.x + threadIdx.x;
  if (i >= NN * 96) return;
  int row = i / 96, c = i % 96;
  float v = 0.f;
  if (c < 6)       v = x[row * 7 + (c < 3 ? c : c + 1)];   // keep = {0,1,2,4,5,6}
  else if (c == 6) v = ini[row];
  hAh[i] = (f16)v;
}

__global__ void k_scatter_t(const float* __restrict__ t, const int* __restrict__ sidx,
                            f16* __restrict__ hAh) {
  int i = blockIdx.x * blockDim.x + threadIdx.x;
  if (i >= NB * 64) return;
  int row = i / 64, c = i % 64;
  hAh[(size_t)sidx[row] * 96 + 7 + c] = (f16)t[i];
}

// ---------------- weight pack: W(Dx64 f32) -> WMMA-B lane layout (f16) ----------------
// Flat index: (((kt*4)+c)*32 + lane)*16 + e ; element = W[kt*32+(lane>>4)*8+(e&7)+(e>>3)*16][c*16+(lane&15)]
__global__ void k_pack_w(const float* __restrict__ W, f16* __restrict__ Wpk,
                         int D, int KP) {
  int i = blockIdx.x * blockDim.x + threadIdx.x;
  if (i >= KP * 64) return;
  int e = i & 15, lane = (i >> 4) & 31, c = (i >> 9) & 3, kt = i >> 11;
  int k = kt * 32 + (lane >> 4) * 8 + (e & 7) + (e >> 3) * 16;
  int col = c * 16 + (lane & 15);
  Wpk[i] = (k < D) ? (f16)W[k * 64 + col] : (f16)0.f;
}

__global__ void k_layer_init(float* __restrict__ agg, float* __restrict__ ssum,
                             float* __restrict__ mmax, float* __restrict__ sums) {
  int i = blockIdx.x * blockDim.x + threadIdx.x;
  if (i < NN * 64) agg[i] = 0.f;
  if (i < NN) { ssum[i] = 0.f; mmax[i] = -INFINITY; }
  if (i < 128) sums[i] = 0.f;
}

// ---------------- node GEMM: hWh = hAh(f16,KP) x Wpk + bias, f16 out ----------------
// One wave per 16-row tile, 4 accumulators = 64 cols, fully unrolled K.
// Fragment layouts per CDNA5 ISA 7.12.2 (wave32 16x16x32 f16 WMMA):
//   A: lane<16 holds K {0..7,16..23}, lane>=16 holds K {8..15,24..31} (row M=lane&15)
//   B: pre-packed so each lane's v16h is one contiguous 32B load
//   C/D: VGPR r -> M = r + 8*(lane>>4), N = lane&15
template <int NKT>
__global__ void k_gemm_wmma(const f16* __restrict__ A, const f16* __restrict__ Wpk,
                            const float* __restrict__ bias, f16* __restrict__ outh) {
  const int lane = threadIdx.x & 31;
  const int tile = blockIdx.x * (blockDim.x >> 5) + (threadIdx.x >> 5);
  if (tile >= NN / 16) return;                  // whole-wave uniform exit
  const int half = lane >> 4;
  const int mloc = lane & 15;
  const int KP = NKT * 32;
  const f16* arow = A + (size_t)(tile * 16 + mloc) * KP + half * 8;
  v8f acc[4] = {v8f{}, v8f{}, v8f{}, v8f{}};
#pragma unroll
  for (int kt = 0; kt < NKT; ++kt) {
    f16x8 g0 = *(const f16x8*)(arow + kt * 32);
    f16x8 g1 = *(const f16x8*)(arow + kt * 32 + 16);
    v16h a = __builtin_shufflevector(g0, g1, 0, 1, 2, 3, 4, 5, 6, 7,
                                     8, 9, 10, 11, 12, 13, 14, 15);
#pragma unroll
    for (int c = 0; c < 4; ++c) {
      v16h b = *(const v16h*)(Wpk + (size_t)(((kt * 4) + c) * 32 + lane) * 16);
      acc[c] = __builtin_amdgcn_wmma_f32_16x16x32_f16(
          false, a, false, b, (short)0, acc[c], false, false);
    }
  }
#pragma unroll
  for (int c = 0; c < 4; ++c) {
    int col = c * 16 + mloc;
    float bc = bias[col];
#pragma unroll
    for (int r = 0; r < 8; ++r) {
      int m = tile * 16 + r + 8 * half;
      outh[(size_t)m * 64 + col] = (f16)(acc[c][r] + bc);
    }
  }
}

// ---------------- edge pass A: z = hWh[src]*sigmoid(e), alpha, segment max ----------------
// 8 edges per 256-thread block; lane handles features {2*lane, 2*lane+1} (one dword).
__global__ void k_edge_attn(const f16* __restrict__ hWh, const int* __restrict__ ei,
                            const float* __restrict__ ea, const float* __restrict__ We,
                            const float* __restrict__ be, const float* __restrict__ attn,
                            f16* __restrict__ z, float* __restrict__ alpha,
                            float* __restrict__ mmax) {
  __shared__ float sWe[192], sbe[64], sattn[64];
  for (int i = threadIdx.x; i < 192; i += 256) sWe[i] = We[i];
  for (int i = threadIdx.x; i < 64; i += 256) { sbe[i] = be[i]; sattn[i] = attn[i]; }
  __syncthreads();
  int lane = threadIdx.x & 31;
  int e = blockIdx.x * 8 + (threadIdx.x >> 5);
  if (e < NE) {
    int src = ei[e], dst = ei[NE + e];
    float a0 = ea[e * 3], a1 = ea[e * 3 + 1], a2 = ea[e * 3 + 2];
    int j0 = 2 * lane, j1 = 2 * lane + 1;
    f16x2 hv = *(const f16x2*)(hWh + (size_t)src * 64 + j0);
    float ev0 = sbe[j0] + a0 * sWe[j0] + a1 * sWe[64 + j0] + a2 * sWe[128 + j0];
    float ev1 = sbe[j1] + a0 * sWe[j1] + a1 * sWe[64 + j1] + a2 * sWe[128 + j1];
    float z0 = (float)hv[0] * (1.f / (1.f + __expf(-ev0)));
    float z1 = (float)hv[1] * (1.f / (1.f + __expf(-ev1)));
    f16x2 zv; zv[0] = (f16)z0; zv[1] = (f16)z1;
    *(f16x2*)(z + (size_t)e * 64 + j0) = zv;
    float al = z0 * sattn[j0] + z1 * sattn[j1];
#pragma unroll
    for (int o = 16; o > 0; o >>= 1) al += __shfl_xor(al, o, 32);
    al = leakyf(al, 0.1f);
    if (lane == 0) { alpha[e] = al; atomicMaxF(&mmax[dst], al); }
  }
}

// ---------------- edge pass B: a = exp(alpha - m[dst]); segment sum ----------------
__global__ void k_edge_exp(const int* __restrict__ ei, float* __restrict__ alpha,
                           const float* __restrict__ mmax, float* __restrict__ ssum) {
  int e = blockIdx.x * blockDim.x + threadIdx.x;
  if (e >= NE) return;
  int dst = ei[NE + e];
  float a = __expf(alpha[e] - mmax[dst]);
  alpha[e] = a;
  atomicAdd(&ssum[dst], a);
}

// ---------------- edge pass C: agg[dst] += z * a/(s+eps) ----------------
__global__ void k_edge_agg(const int* __restrict__ ei, const float* __restrict__ alpha,
                           const float* __restrict__ ssum, const f16* __restrict__ z,
                           float* __restrict__ agg) {
  int lane = threadIdx.x & 31;
  int e = blockIdx.x * 8 + (threadIdx.x >> 5);
  if (e >= NE) return;
  int dst = ei[NE + e];
  float w = alpha[e] / (ssum[dst] + 1e-16f);
  int j0 = 2 * lane;
  f16x2 zv = *(const f16x2*)(z + (size_t)e * 64 + j0);
  atomicAdd(&agg[(size_t)dst * 64 + j0],     (float)zv[0] * w);
  atomicAdd(&agg[(size_t)dst * 64 + j0 + 1], (float)zv[1] * w);
}

// ---------------- GraphNorm: per-feature sum / sumsq reduction ----------------
__global__ void k_gn_reduce(const float* __restrict__ agg, float* __restrict__ sums) {
  __shared__ float sh[512];
  int f = threadIdx.x & 63, g = threadIdx.x >> 6;
  int r0 = blockIdx.x * 64;
  float s = 0.f, q = 0.f;
  for (int r = g; r < 64; r += 4) {
    int node = r0 + r;
    if (node < NN) { float v = agg[(size_t)node * 64 + f]; s += v; q += v * v; }
  }
  sh[threadIdx.x] = s; sh[256 + threadIdx.x] = q;
  __syncthreads();
  if (threadIdx.x < 64) {
    atomicAdd(&sums[f],      sh[f] + sh[64 + f] + sh[128 + f] + sh[192 + f]);
    atomicAdd(&sums[64 + f], sh[256 + f] + sh[320 + f] + sh[384 + f] + sh[448 + f]);
  }
}

__global__ void k_gn_finalize(const float* __restrict__ sums, const float* __restrict__ w,
                              const float* __restrict__ b, const float* __restrict__ ms,
                              float* __restrict__ sc) {
  int f = threadIdx.x;
  if (f >= 64) return;
  float mean = sums[f] * (1.f / NN);
  float mu = mean * ms[f];
  float var = sums[64 + f] * (1.f / NN) - 2.f * mu * mean + mu * mu;
  float inv = rsqrtf(var + 1e-5f);
  sc[f] = w[f] * inv;
  sc[64 + f] = b[f] - mu * w[f] * inv;
}

// apply affine + leaky; produce f16 features for the next GEMM and f32 for readout
__global__ void k_gn_apply(const float* __restrict__ agg, const float* __restrict__ sc,
                           f16* __restrict__ hAh, float* __restrict__ hA32) {
  int i = blockIdx.x * blockDim.x + threadIdx.x;
  if (i >= NN * 64) return;
  int f = i & 63;
  float v = leakyf(sc[f] * agg[i] + sc[64 + f], 0.1f);
  hAh[i] = (f16)v;
  hA32[i] = v;
}

// ---------------- readout head ----------------
__global__ void k_readout(const float* __restrict__ h, const float* __restrict__ W1,
                          const float* __restrict__ b1, const float* __restrict__ W2,
                          const float* __restrict__ b2, const float* __restrict__ ini,
                          float* __restrict__ out) {
  int lane = threadIdx.x & 31;
  int n = blockIdx.x * 8 + (threadIdx.x >> 5);
  if (n >= NN) return;
  const float* hr = h + (size_t)n * 64;
  float r = b1[lane];
#pragma unroll 8
  for (int k = 0; k < 64; ++k) r += hr[k] * W1[k * 32 + lane];
  r = leakyf(r, 0.1f);
  float acc = r * W2[lane];
#pragma unroll
  for (int o = 16; o > 0; o >>= 1) acc += __shfl_xor(acc, o, 32);
  if (lane == 0) out[n] = ini[n] + tanhf(acc + b2[0]);
}

// ---------------------------------------------------------------------------
extern "C" void kernel_launch(void* const* d_in, const int* in_sizes, int n_in,
                              void* d_out, int out_size, void* d_ws, size_t ws_size,
                              hipStream_t stream) {
  (void)in_sizes; (void)n_in; (void)out_size; (void)ws_size;
  const float* x    = (const float*)d_in[0];
  const float* ini  = (const float*)d_in[1];
  const float* xb   = (const float*)d_in[2];
  const int*   sidx = (const int*)d_in[3];
  const int*   ei   = (const int*)d_in[4];
  const float* ea   = (const float*)d_in[5];
  const float* tW1 = (const float*)d_in[6],  *tb1 = (const float*)d_in[7];
  const float* tW2 = (const float*)d_in[8],  *tb2 = (const float*)d_in[9];
  const float* tlw = (const float*)d_in[10], *tlb = (const float*)d_in[11];
  const float *Wn[5], *bn[5], *We[5], *be[5], *attn[5], *gw[5], *gb[5], *gms[5];
  for (int l = 0; l < 5; ++l) {
    Wn[l]   = (const float*)d_in[12 + 5 * l];
    bn[l]   = (const float*)d_in[13 + 5 * l];
    We[l]   = (const float*)d_in[14 + 5 * l];
    be[l]   = (const float*)d_in[15 + 5 * l];
    attn[l] = (const float*)d_in[16 + 5 * l];
    gw[l]   = (const float*)d_in[37 + 3 * l];
    gb[l]   = (const float*)d_in[38 + 3 * l];
    gms[l]  = (const float*)d_in[39 + 3 * l];
  }
  const float* rW1 = (const float*)d_in[52], *rb1 = (const float*)d_in[53];
  const float* rW2 = (const float*)d_in[54], *rb2 = (const float*)d_in[55];
  float* out = (float*)d_out;

  // workspace carve (256B aligned); total ~148 MB
  char* base = (char*)d_ws; size_t off = 0;
  auto carve = [&](size_t bytes) -> char* {
    off = (off + 255) & ~(size_t)255; char* r = base + off; off += bytes; return r;
  };
  f16*   hAh   = (f16*)  carve((size_t)NN * 96 * 2);   // f16 node features (layer in)
  f16*   hWh   = (f16*)  carve((size_t)NN * 64 * 2);   // f16 transformed nodes
  float* hA32  = (float*)carve((size_t)NN * 64 * 4);   // f32 copy for readout
  float* agg   = (float*)carve((size_t)NN * 64 * 4);
  f16*   z     = (f16*)  carve((size_t)NE * 64 * 2);
  float* alpha = (float*)carve((size_t)NE * 4);
  float* mmax  = (float*)carve((size_t)NN * 4);
  float* ssum  = (float*)carve((size_t)NN * 4);
  f16*   Wpk   = (f16*)  carve((size_t)96 * 64 * 2);
  float* t1    = (float*)carve((size_t)NB * 96 * 4);
  float* t2    = (float*)carve((size_t)NB * 64 * 4);
  float* sums  = (float*)carve(128 * 4);
  float* sc    = (float*)carve(128 * 4);

  const int TB = 256;
  // branch MLP + feature assembly
  k_bmlp1<<<(NB * 96 + TB - 1) / TB, TB, 0, stream>>>(xb, tW1, tb1, t1);
  k_bmlp2<<<(NB * 64 + TB - 1) / TB, TB, 0, stream>>>(t1, tW2, tb2, t2);
  k_bln  <<<(NB + TB - 1) / TB, TB, 0, stream>>>(t2, tlw, tlb);
  k_fill_h0<<<(NN * 96 + TB - 1) / TB, TB, 0, stream>>>(x, ini, hAh);
  k_scatter_t<<<(NB * 64 + TB - 1) / TB, TB, 0, stream>>>(t2, sidx, hAh);

  // 5 GAT layers
  for (int l = 0; l < 5; ++l) {
    const int D  = (l == 0) ? 71 : 64;
    const int KP = (l == 0) ? 96 : 64;
    k_pack_w<<<(KP * 64 + TB - 1) / TB, TB, 0, stream>>>(Wn[l], Wpk, D, KP);
    k_layer_init<<<(NN * 64 + TB - 1) / TB, TB, 0, stream>>>(agg, ssum, mmax, sums);
    if (l == 0)
      k_gemm_wmma<3><<<(NN / 16 + 7) / 8, TB, 0, stream>>>(hAh, Wpk, bn[l], hWh);
    else
      k_gemm_wmma<2><<<(NN / 16 + 7) / 8, TB, 0, stream>>>(hAh, Wpk, bn[l], hWh);
    k_edge_attn<<<(NE + 7) / 8, TB, 0, stream>>>(hWh, ei, ea, We[l], be[l], attn[l],
                                                 z, alpha, mmax);
    k_edge_exp<<<(NE + TB - 1) / TB, TB, 0, stream>>>(ei, alpha, mmax, ssum);
    k_edge_agg<<<(NE + 7) / 8, TB, 0, stream>>>(ei, alpha, ssum, z, agg);
    k_gn_reduce<<<(NN + 63) / 64, TB, 0, stream>>>(agg, sums);
    k_gn_finalize<<<1, 64, 0, stream>>>(sums, gw[l], gb[l], gms[l], sc);
    k_gn_apply<<<(NN * 64 + TB - 1) / TB, TB, 0, stream>>>(agg, sc, hAh, hA32);
  }

  // readout
  k_readout<<<NN / 8, TB, 0, stream>>>(hA32, rW1, rb1, rW2, rb2, ini, out);
}
